// Attention_14431090114722
// MI455X (gfx1250) — matrix operand order repaired and checked by reading, not compiled, not run on hardware
//
#include <hip/hip_runtime.h>
#include <hip/hip_bf16.h>

typedef __attribute__((ext_vector_type(16))) _Float16 v16h;
typedef __attribute__((ext_vector_type(8)))  _Float16 v8h;
typedef __attribute__((ext_vector_type(8)))  float    v8f;

#define BATCH 8
#define CTX   2048
#define DIM   1024
#define MTOT  (BATCH * CTX)   // 16384

__device__ __forceinline__ v16h ld16(const _Float16* p0, const _Float16* p1) {
    v8h a = *(const v8h*)p0;
    v8h b = *(const v8h*)p1;
    v16h r;
#pragma unroll
    for (int i = 0; i < 8; ++i) { r[i] = a[i]; r[i + 8] = b[i]; }
    return r;
}

__device__ __forceinline__ v8f wmma16(v16h a, v16h b, v8f c) {
    return __builtin_amdgcn_wmma_f32_16x16x32_f16(false, a, false, b, (short)0, c,
                                                  false, false);
}

// ---------------------------------------------------------------- pack x -> f16
__global__ __launch_bounds__(256) void pack_x_kernel(const float* __restrict__ x,
                                                     _Float16* __restrict__ xh) {
    size_t i = ((size_t)blockIdx.x * blockDim.x + threadIdx.x) * 8;
    const float4* p = (const float4*)(x + i);
    float4 a = p[0], b = p[1];
    v8h o;
    o[0] = (_Float16)a.x; o[1] = (_Float16)a.y; o[2] = (_Float16)a.z; o[3] = (_Float16)a.w;
    o[4] = (_Float16)b.x; o[5] = (_Float16)b.y; o[6] = (_Float16)b.z; o[7] = (_Float16)b.w;
    *(v8h*)(xh + i) = o;
}

// ------------------------------------------- W[i][o] fp32 -> Wt[o][i] f16 (LDS tiled)
__global__ __launch_bounds__(256) void pack_w_kernel(const float* __restrict__ W,
                                                     _Float16* __restrict__ Wt) {
    __shared__ float tile[32][33];
    int i0 = blockIdx.x * 32, o0 = blockIdx.y * 32;
    int tx = threadIdx.x & 31, ty = threadIdx.x >> 5; // 8 row-groups
#pragma unroll
    for (int r = 0; r < 32; r += 8)
        tile[r + ty][tx] = W[(size_t)(i0 + r + ty) * DIM + o0 + tx];
    __syncthreads();
#pragma unroll
    for (int r = 0; r < 32; r += 8)
        Wt[(size_t)(o0 + r + ty) * DIM + i0 + tx] = (_Float16)tile[tx][r + ty];
}

// ------------------------------------------------------- projection GEMM (WMMA)
// C[m, n] = sum_k xh[m, k] * wt[n, k]. Each wave: one 16x64 tile.
// TRANS=false: out row-major f16 [MTOT][DIM]. TRANS=true: out = Vt[b][d][s].
template <bool TRANS>
__global__ __launch_bounds__(128) void proj_gemm_kernel(const _Float16* __restrict__ xh,
                                                        const _Float16* __restrict__ wt,
                                                        _Float16* __restrict__ out) {
    int wave = blockIdx.x * (blockDim.x >> 5) + (threadIdx.x >> 5);
    const int NT = DIM / 64;              // 16 n-tiles
    int m0 = (wave / NT) * 16;
    int n0 = (wave % NT) * 64;
    int lane = threadIdx.x & 31;
    int h = lane >> 4, r = lane & 15;

    v8f acc[4] = {};
    const _Float16* arow = xh + (size_t)(m0 + r) * DIM;
    for (int kb = 0; kb < DIM; kb += 32) {
        v16h a = ld16(arow + kb + h * 8, arow + kb + 16 + h * 8);
#pragma unroll
        for (int t = 0; t < 4; ++t) {
            const _Float16* brow = wt + (size_t)(n0 + t * 16 + r) * DIM + kb + h * 8;
            v16h b = ld16(brow, brow + 16);
            acc[t] = wmma16(a, b, acc[t]);
        }
    }
    if (TRANS) {
        int bb = m0 / CTX, s0 = m0 % CTX;
#pragma unroll
        for (int t = 0; t < 4; ++t) {
            v8h o;
#pragma unroll
            for (int j = 0; j < 8; ++j) o[j] = (_Float16)acc[t][j];
            size_t d = (size_t)bb * DIM + n0 + t * 16 + r;
            *(v8h*)(out + d * CTX + s0 + 8 * h) = o;
        }
    } else {
#pragma unroll
        for (int t = 0; t < 4; ++t)
#pragma unroll
            for (int j = 0; j < 8; ++j)
                out[(size_t)(m0 + 8 * h + j) * DIM + n0 + t * 16 + r] = (_Float16)acc[t][j];
    }
}

// --------------------------------------------------------- flash attention (WMMA)
// One 256-thread WG per (batch, 16-row query tile). kv step 256.
__global__ __launch_bounds__(256) void flash_attn_kernel(const _Float16* __restrict__ Qh,
                                                         const _Float16* __restrict__ Kh,
                                                         const _Float16* __restrict__ Vt,
                                                         float* __restrict__ out) {
    __shared__ __align__(16) _Float16 Qs[16 * DIM];     // 32 KB
    __shared__ __align__(16) float    Smat[16 * 256];   // 16 KB
    __shared__ __align__(16) _Float16 Pmat[16 * 256];   //  8 KB
    __shared__ float red[16 * 16];
    __shared__ float mrow[16], lrow[16], corr[16];

    const int tile = blockIdx.x;
    const int b  = tile / (CTX / 16);
    const int q0 = (tile % (CTX / 16)) * 16;
    const int tid  = threadIdx.x;
    const int wid  = tid >> 5;
    const int lane = tid & 31;
    const int h = lane >> 4, r = lane & 15;
    const int d0 = wid * 128;
    const float scale = 0.03125f;  // 1/sqrt(1024)

    // stage Q tile (16 contiguous rows) into LDS
    const _Float16* Qg = Qh + ((size_t)b * CTX + q0) * DIM;
    for (int i = tid * 8; i < 16 * DIM; i += 256 * 8)
        *(v8h*)(Qs + i) = *(const v8h*)(Qg + i);
    if (tid < 16) { mrow[tid] = -3.0e38f; lrow[tid] = 0.0f; }

    v8f acc[8] = {};
    const int kv_iters = (q0 + 16 + 255) >> 8;
    __syncthreads();

    for (int it = 0; it < kv_iters; ++it) {
        const int kv0 = it << 8;
        // ---- scores: this wave computes S for kv columns [wid*32, wid*32+32)
        v8f s0 = {}, s1 = {};
        {
            const _Float16* k0 = Kh + ((size_t)b * CTX + kv0 + wid * 32 + r) * DIM;
            const _Float16* k1 = k0 + (size_t)16 * DIM;
            __builtin_prefetch(k0, 0, 1);
            for (int kb = 0; kb < DIM; kb += 32) {
                v16h a  = ld16(Qs + r * DIM + kb + h * 8, Qs + r * DIM + kb + 16 + h * 8);
                v16h b0 = ld16(k0 + kb + h * 8, k0 + kb + 16 + h * 8);
                s0 = wmma16(a, b0, s0);
                v16h b1 = ld16(k1 + kb + h * 8, k1 + kb + 16 + h * 8);
                s1 = wmma16(a, b1, s1);
            }
        }
        // ---- scaled + causal-masked scores -> LDS
#pragma unroll
        for (int j = 0; j < 8; ++j) {
            int m = 8 * h + j;
            int q_abs = q0 + m;
            int c0 = wid * 32 + r;
            int c1 = c0 + 16;
            Smat[m * 256 + c0] = (kv0 + c0 <= q_abs) ? s0[j] * scale : -3.0e38f;
            Smat[m * 256 + c1] = (kv0 + c1 <= q_abs) ? s1[j] * scale : -3.0e38f;
        }
        __syncthreads();
        // ---- row max (cooperative): thread = (row rr, strided cols cc)
        const int rr = tid >> 4, cc = tid & 15;
        float pm = -3.0e38f;
        for (int c = cc; c < 256; c += 16) pm = fmaxf(pm, Smat[rr * 256 + c]);
        red[rr * 16 + cc] = pm;
        __syncthreads();
        if (tid < 16) {
            float m_old = mrow[tid], mx = m_old;
#pragma unroll
            for (int c = 0; c < 16; ++c) mx = fmaxf(mx, red[tid * 16 + c]);
            float cf = (m_old <= -1.0e38f) ? 0.0f : __expf(m_old - mx);
            mrow[tid] = mx; corr[tid] = cf; lrow[tid] *= cf;
        }
        __syncthreads();
        // ---- P = exp(S - m_new) -> LDS (f16), partial row sums
        float mnew = mrow[rr];
        float psum = 0.0f;
        for (int c = cc; c < 256; c += 16) {
            float p = __expf(Smat[rr * 256 + c] - mnew);
            Pmat[rr * 256 + c] = (_Float16)p;
            psum += p;
        }
        red[rr * 16 + cc] = psum;
        __syncthreads();
        if (tid < 16) {
            float s = 0.0f;
#pragma unroll
            for (int c = 0; c < 16; ++c) s += red[tid * 16 + c];
            lrow[tid] += s;
        }
        // ---- rescale running accumulator by correction factor
        float cf8[8];
#pragma unroll
        for (int j = 0; j < 8; ++j) cf8[j] = corr[8 * h + j];
#pragma unroll
        for (int dt = 0; dt < 8; ++dt)
#pragma unroll
            for (int j = 0; j < 8; ++j) acc[dt][j] *= cf8[j];
        // ---- P @ V for this wave's 128-wide d slice
        const _Float16* vbase = Vt + (size_t)b * DIM * CTX + kv0;
        for (int kt = 0; kt < 8; ++kt) {
            v16h a = ld16(Pmat + r * 256 + kt * 32 + h * 8,
                          Pmat + r * 256 + kt * 32 + 16 + h * 8);
#pragma unroll
            for (int dt = 0; dt < 8; ++dt) {
                const _Float16* vp = vbase + (size_t)(d0 + dt * 16 + r) * CTX + kt * 32 + h * 8;
                v16h bb = ld16(vp, vp + 16);
                acc[dt] = wmma16(a, bb, acc[dt]);
            }
        }
        __syncthreads();
    }

    // ---- normalize and store fp32 output
    float linv[8];
#pragma unroll
    for (int j = 0; j < 8; ++j) {
        float l = lrow[8 * h + j];
        linv[j] = (l > 0.0f) ? 1.0f / l : 0.0f;
    }
    float* og = out + ((size_t)b * CTX + q0) * DIM + d0;
#pragma unroll
    for (int dt = 0; dt < 8; ++dt)
#pragma unroll
        for (int j = 0; j < 8; ++j)
            og[(size_t)(8 * h + j) * DIM + dt * 16 + r] = acc[dt][j] * linv[j];
}

// ------------------------------------------------------------------- launcher
extern "C" void kernel_launch(void* const* d_in, const int* in_sizes, int n_in,
                              void* d_out, int out_size, void* d_ws, size_t ws_size,
                              hipStream_t stream) {
    const float* x  = (const float*)d_in[0];
    const float* Wq = (const float*)d_in[1];
    const float* Wk = (const float*)d_in[2];
    const float* Wv = (const float*)d_in[3];
    float* out = (float*)d_out;

    // workspace layout (f16): xh | wtq wtk wtv | Qh | Kh | Vt
    _Float16* xh  = (_Float16*)d_ws;
    _Float16* wtq = xh  + (size_t)MTOT * DIM;
    _Float16* wtk = wtq + (size_t)DIM * DIM;
    _Float16* wtv = wtk + (size_t)DIM * DIM;
    _Float16* Qh  = wtv + (size_t)DIM * DIM;
    _Float16* Kh  = Qh  + (size_t)MTOT * DIM;
    _Float16* Vt  = Kh  + (size_t)MTOT * DIM;

    // 1) pack inputs to f16
    {
        size_t n = (size_t)MTOT * DIM;                 // 16.7M elements
        dim3 grid((unsigned)(n / (256 * 8)));
        pack_x_kernel<<<grid, 256, 0, stream>>>(x, xh);
    }
    {
        dim3 grid(DIM / 32, DIM / 32);
        pack_w_kernel<<<grid, 256, 0, stream>>>(Wq, wtq);
        pack_w_kernel<<<grid, 256, 0, stream>>>(Wk, wtk);
        pack_w_kernel<<<grid, 256, 0, stream>>>(Wv, wtv);
    }
    // 2) projections (Q, K row-major; V transposed)
    {
        int waves = (MTOT / 16) * (DIM / 64);          // 16384
        dim3 grid(waves / 4);
        proj_gemm_kernel<false><<<grid, 128, 0, stream>>>(xh, wtq, Qh);
        proj_gemm_kernel<false><<<grid, 128, 0, stream>>>(xh, wtk, Kh);
        proj_gemm_kernel<true ><<<grid, 128, 0, stream>>>(xh, wtv, Vt);
    }
    // 3) flash attention
    {
        dim3 grid(BATCH * (CTX / 16));                 // 1024
        flash_attn_kernel<<<grid, 256, 0, stream>>>(Qh, Kh, Vt, out);
    }
}
